// Multi_Head_Attention_89395449299118
// MI455X (gfx1250) — matrix operand order repaired and checked by reading, not compiled, not run on hardware
//
#include <hip/hip_runtime.h>

// ---------------------------------------------------------------------------
// MHA forward on gfx1250: bf16 WMMA (v_wmma_f32_16x16x32_bf16) everywhere,
// with TDM (tensor_load_to_lds) double-buffered staging of ALL streamed
// operands (GEMM A+B slabs, flash-attention K^T and V tiles, shared by the
// 4 waves of each block). Shapes hardcoded: B=2, S=2048, D=2048, H=16, hd=128.
// ---------------------------------------------------------------------------

typedef __attribute__((ext_vector_type(16))) __bf16 v16bf;
typedef __attribute__((ext_vector_type(8)))  float  v8f;
typedef unsigned int u32x4 __attribute__((ext_vector_type(4)));
typedef int          i32x8 __attribute__((ext_vector_type(8)));
typedef int          i32x4 __attribute__((ext_vector_type(4)));

#if __has_builtin(__builtin_amdgcn_tensor_load_to_lds)
#define HAVE_TDM 1
#else
#define HAVE_TDM 0
#endif

static constexpr int Bc  = 2;
static constexpr int Sc  = 2048;
static constexpr int Dc  = 2048;
static constexpr int Hc  = 16;
static constexpr int HDc = 128;               // head dim
static constexpr int Mc  = Bc * Sc;           // 4096 rows

// Padded LDS row strides (TDM pad inserts 4 DWORDs per row to break bank
// conflict patterns on the ds_load_b128 fragment reads):
static constexpr int AROW = 40;   // 32 bf16 row  -> 80 B   (pad code 3/3)
static constexpr int BROW = 72;   // 64 bf16 row  -> 144 B  (pad code 4/3)
static constexpr int KROW = 40;   // 32 bf16 row  -> 80 B   (pad code 3/3)
static constexpr int VROW = 136;  // 128 bf16 row -> 272 B  (pad code 5/3)

static constexpr int ATILE_USH = 64 * AROW;   // 5120 B
static constexpr int BTILE_USH = 32 * BROW;   // 4608 B
static constexpr int KTILE_USH = 128 * KROW;  // 10240 B
static constexpr int VTILE_USH = 32 * VROW;   // 8704 B

union FragBF {
    v16bf v;
    uint4 q[2];
};

// fp32 -> bf16, round to nearest even
__device__ __forceinline__ unsigned short f2bf(float f) {
    unsigned int u = __float_as_uint(f);
    unsigned int r = (u + 0x7FFFu + ((u >> 16) & 1u)) >> 16;
    return (unsigned short)r;
}

// A fragment (16x32 bf16, MxK). ISA layout: lanes 0-15 hold rows M=0..15 with
// K=0..7 (VGPR0-3) and K=16..23 (VGPR4-7); lanes 16-31 hold the same rows with
// K=8..15 and K=24..31.
__device__ __forceinline__ v16bf load_a_frag(const unsigned short* base, int ld,
                                             int row0, int k0) {
    const int lane = threadIdx.x & 31;
    const int r    = row0 + (lane & 15);
    const int sel  = lane >> 4;
    const unsigned short* p = base + (size_t)r * ld + k0 + sel * 8;
    FragBF f;
    f.q[0] = *(const uint4*)p;          // K = sel*8 .. sel*8+7
    f.q[1] = *(const uint4*)(p + 16);   // K = 16+sel*8 .. 16+sel*8+7
    return f.v;
}

// B fragment (32x16 bf16, KxN). Lane l holds column N=l%16 with K=8h..8h+7
// and K=16+8h..16+8h+7 (h=l/16), matching the A fragment K order.
__device__ __forceinline__ v16bf load_b_frag(const unsigned short* base, int ld,
                                             int k0, int n0) {
    const int lane = threadIdx.x & 31;
    const unsigned short* p = base + (size_t)(k0 + (lane >> 4) * 8) * ld + n0 + (lane & 15);
    unsigned short t[16];
#pragma unroll
    for (int s = 0; s < 8; ++s) {
        t[s]     = p[(size_t)s * ld];
        t[s + 8] = p[(size_t)(s + 16) * ld];
    }
    FragBF f;
    __builtin_memcpy(&f, t, 32);
    return f.v;
}

__device__ __forceinline__ v8f wmma_bf16(v16bf a, v16bf b, v8f c) {
    return __builtin_amdgcn_wmma_f32_16x16x32_bf16(false, a, false, b,
                                                   (short)0, c, false, false);
}

#if HAVE_TDM
// Issue a TDM 2D tile load of (tile_h rows x tile_w bf16) from a row-major
// matrix with row stride stride_elems into LDS at byte offset lds_off.
// pad_i/pad_a are the D# pad_interval / pad_amount codes (pad after each row).
__device__ __forceinline__ void tdm_issue_2d(const unsigned short* gptr,
                                             unsigned lds_off,
                                             int tile_w, int tile_h,
                                             int stride_elems,
                                             unsigned pad_i, unsigned pad_a) {
    unsigned long long ga = (unsigned long long)(size_t)gptr;
    u32x4 g0;
    g0[0] = 1u;                                     // count=1 (valid descriptor)
    g0[1] = lds_off;                                // lds_addr (bytes)
    g0[2] = (unsigned)ga;                           // global_addr[31:0]
    g0[3] = (unsigned)((ga >> 32) & 0x01FFFFFFull)  // global_addr[56:32]
          | 0x80000000u;                            // type=2 ("image")
    i32x8 g1;
    g1[0] = (int)((1u << 16)        // data_size = 1 -> 2 bytes
                | (1u << 20)        // pad_enable
                | (pad_i << 22)     // pad_interval code
                | (pad_a << 25));   // pad_amount code
    g1[1] = (int)(((unsigned)stride_elems & 0xFFFFu) << 16);  // tensor_dim0 lo16
    g1[2] = (int)(((unsigned)stride_elems >> 16) & 0xFFFFu);  // dim0 hi | dim1 lo(0)
    g1[3] = (int)(0x10u | ((unsigned)tile_w << 16));          // dim1 hi (=1<<20) | tile_dim0
    g1[4] = tile_h;                                           // tile_dim1
    g1[5] = stride_elems;                                     // tensor_dim0_stride
    g1[6] = 0;
    g1[7] = 0;
    i32x4 gz4 = {0, 0, 0, 0};
    i32x8 gz8 = {0, 0, 0, 0, 0, 0, 0, 0};
    __builtin_amdgcn_tensor_load_to_lds(g0, g1, gz4, gz4, gz8, 0);
}
#endif

// ---------------------------------------------------------------------------
// Kernel 1: fp32 -> bf16 cast (vectorized x4)
// ---------------------------------------------------------------------------
__global__ void cast_f32_to_bf16(const float* __restrict__ in,
                                 unsigned short* __restrict__ out, int n) {
    int i = (blockIdx.x * blockDim.x + threadIdx.x) * 4;
    if (i + 3 < n) {
        float4 f = *(const float4*)(in + i);
        ushort4 u;
        u.x = f2bf(f.x); u.y = f2bf(f.y); u.z = f2bf(f.z); u.w = f2bf(f.w);
        *(ushort4*)(out + i) = u;
    }
}

// ---------------------------------------------------------------------------
// Kernel 2: WT[k][n] = bf16(W[n][k])  (32x32 LDS tile transpose), D x D
// ---------------------------------------------------------------------------
__global__ void transpose_cast(const float* __restrict__ W,
                               unsigned short* __restrict__ WT) {
    __shared__ float tile[32][33];
    const int bx = blockIdx.x * 32;   // k-block
    const int by = blockIdx.y * 32;   // n-block
    const int tx = threadIdx.x, ty = threadIdx.y;   // 32 x 8
#pragma unroll
    for (int i = ty; i < 32; i += 8)
        tile[i][tx] = W[(size_t)(by + i) * Dc + bx + tx];
    __syncthreads();
#pragma unroll
    for (int i = ty; i < 32; i += 8)
        WT[(size_t)(bx + i) * Dc + by + tx] = f2bf(tile[tx][i]);
}

// ---------------------------------------------------------------------------
// Kernel 3: C = A(bf16, MxK row-major) * Bt(bf16, KxN row-major), fp32 accum.
// 128 threads (4 waves) per block computing a 64x64 tile (wave w owns rows
// row0+16w). BOTH the 64x32 A slab and the 32x64 B slab are staged into LDS
// by TDM, double buffered: the two DMAs for k-slab i+1 run while WMMAs
// consume slab i entirely out of LDS. Store modes:
//   0: bf16 row-major C[m][n]
//   1: bf16 transposed per head:  Kt[((b*H+h)*hd + d) * S + s]
//   2: fp32 row-major C[m][n]
// ---------------------------------------------------------------------------
__global__ void gemm_bf16(const unsigned short* __restrict__ A,
                          const unsigned short* __restrict__ Bt,
                          void* __restrict__ Cout, int mode) {
    __shared__ __attribute__((aligned(128))) unsigned short Alds[2][ATILE_USH];
    __shared__ __attribute__((aligned(128))) unsigned short Blds[2][BTILE_USH];

    const int tid    = threadIdx.x;
    const int wave   = tid >> 5;
    const int lane   = tid & 31;
    const int halfl  = lane >> 4;
    const int nl     = lane & 15;
    const int n0     = blockIdx.x * 64;
    const int rowblk = blockIdx.y * 64;
    const int row0   = rowblk + wave * 16;

    auto stage_issue = [&](int k0, int buf) {
#if HAVE_TDM
        if (wave == 0) {
            tdm_issue_2d(A + (size_t)rowblk * Dc + k0,
                         (unsigned)(size_t)&Alds[buf][0],
                         /*w*/32, /*h*/64, Dc, /*pad_i*/3u, /*pad_a*/3u);
            tdm_issue_2d(Bt + (size_t)k0 * Dc + n0,
                         (unsigned)(size_t)&Blds[buf][0],
                         /*w*/64, /*h*/32, Dc, /*pad_i*/4u, /*pad_a*/3u);
        }
#else
        {   // A: 64 rows x 32 -> two threads per row
            const int r   = tid >> 1;
            const int seg = (tid & 1) * 16;
            const unsigned short* src = A + (size_t)(rowblk + r) * Dc + k0 + seg;
            unsigned short* dst = &Alds[buf][r * AROW + seg];
            uint4 v0 = *(const uint4*)src;
            uint4 v1 = *(const uint4*)(src + 8);
            *(uint4*)dst = v0;
            *(uint4*)(dst + 8) = v1;
        }
        {   // B: 32 rows x 64 -> four threads per row
            const int r   = tid >> 2;
            const int seg = (tid & 3) * 16;
            const unsigned short* src = Bt + (size_t)(k0 + r) * Dc + n0 + seg;
            unsigned short* dst = &Blds[buf][r * BROW + seg];
            uint4 v0 = *(const uint4*)src;
            uint4 v1 = *(const uint4*)(src + 8);
            *(uint4*)dst = v0;
            *(uint4*)(dst + 8) = v1;
        }
#endif
    };
    auto stage_wait = [&]() {
#if HAVE_TDM
        if (wave == 0) __builtin_amdgcn_s_wait_tensorcnt(0);
#endif
    };

    v8f acc[4];
#pragma unroll
    for (int t = 0; t < 4; ++t) acc[t] = {};

    constexpr int NK = Dc / 32;          // 64 k-slabs

    stage_issue(0, 0);
    stage_wait();
    __syncthreads();

    for (int ks = 0; ks < NK; ++ks) {
        const int cur = ks & 1;
        if (ks + 1 < NK) stage_issue((ks + 1) * 32, cur ^ 1);

        v16bf bfr[4];
#pragma unroll
        for (int t = 0; t < 4; ++t)
            bfr[t] = load_b_frag(&Blds[cur][0], BROW, 0, t * 16);
        v16bf a = load_a_frag(&Alds[cur][0], AROW, wave * 16, 0);
#pragma unroll
        for (int t = 0; t < 4; ++t)
            acc[t] = wmma_bf16(a, bfr[t], acc[t]);

        if (ks + 1 < NK) stage_wait();
        __syncthreads();
    }

#pragma unroll
    for (int t = 0; t < 4; ++t) {
#pragma unroll
        for (int r = 0; r < 8; ++r) {
            const int m = row0 + r + 8 * halfl;       // C-frag M mapping
            const int n = n0 + 16 * t + nl;           // C-frag N mapping
            const float v = acc[t][r];
            if (mode == 0) {
                ((unsigned short*)Cout)[(size_t)m * Dc + n] = f2bf(v);
            } else if (mode == 1) {
                const int b = m / Sc, s = m % Sc;
                const int h = n / HDc, d = n % HDc;
                ((unsigned short*)Cout)[((size_t)(b * Hc + h) * HDc + d) * Sc + s] = f2bf(v);
            } else {
                ((float*)Cout)[(size_t)m * Dc + n] = v;
            }
        }
    }
}

// ---------------------------------------------------------------------------
// Kernel 4: causal flash attention. 128 threads (4 waves) per block; the
// block owns 64 query rows of one (b,h) pair (wave w -> rows q0+16w..+15).
// One double-buffered K^T tile (128x32) + V tile (32x128) per block is
// TDM-DMA'd by wave 0 and SHARED by all 4 waves (4x less DMA traffic than
// per-wave staging). Waves past their causal diagonal skip compute
// (wave-uniform branch, EXEC stays all-ones for WMMA) but keep barriers.
// Output Ob: bf16 [B,S,D].
// ---------------------------------------------------------------------------
__global__ void flash_attn(const unsigned short* __restrict__ Qb,
                           const unsigned short* __restrict__ Kt,
                           const unsigned short* __restrict__ Vb,
                           unsigned short* __restrict__ Ob) {
    __shared__ __attribute__((aligned(128))) unsigned short Klds[2][KTILE_USH];
    __shared__ __attribute__((aligned(128))) unsigned short Vlds[2][VTILE_USH];
    __shared__ __attribute__((aligned(16)))  unsigned short Plds[4][16 * 32];

    const int tid   = threadIdx.x;
    const int wave  = tid >> 5;
    const int lane  = tid & 31;
    const int halfl = lane >> 4;
    const int nl    = lane & 15;
    const int qblk  = blockIdx.x * 64;      // block's first query row
    const int q0    = qblk + wave * 16;     // this wave's first query row
    const int bh    = blockIdx.y;
    const int b     = bh / Hc;
    const int h     = bh % Hc;

    const unsigned short* Qp  = Qb + (size_t)b * Sc * Dc + (size_t)h * HDc;
    const unsigned short* Ktp = Kt + (size_t)bh * HDc * Sc;
    const unsigned short* Vp  = Vb + (size_t)b * Sc * Dc + (size_t)h * HDc;

    // stage K^T tile (128 x 32 @ j0) and V tile (32 x 128 @ j0) into buf
    auto issue_tiles = [&](int j0, int buf) {
#if HAVE_TDM
        if (wave == 0) {
            tdm_issue_2d(Ktp + j0, (unsigned)(size_t)&Klds[buf][0],
                         /*w*/32, /*h*/128, Sc, /*pad_i*/3u, /*pad_a*/3u);
            tdm_issue_2d(Vp + (size_t)j0 * Dc, (unsigned)(size_t)&Vlds[buf][0],
                         /*w*/128, /*h*/32, Dc, /*pad_i*/5u, /*pad_a*/3u);
        }
#else
        {   // K^T: 128 rows x 32 -> one row per thread
            const int r = tid;
            const unsigned short* src = Ktp + (size_t)r * Sc + j0;
            unsigned short* dst = &Klds[buf][r * KROW];
            *(uint4*)dst       = *(const uint4*)src;
            *(uint4*)(dst + 8) = *(const uint4*)(src + 8);
        }
        {   // V: 32 rows x 128 -> four threads per row
            const int r   = tid >> 2;
            const int seg = (tid & 3) * 32;
            const unsigned short* src = Vp + (size_t)(j0 + r) * Dc + seg;
            unsigned short* dst = &Vlds[buf][r * VROW + seg];
#pragma unroll
            for (int s8 = 0; s8 < 4; ++s8)
                *(uint4*)(dst + s8 * 8) = *(const uint4*)(src + s8 * 8);
        }
#endif
    };
    auto wait_tiles = [&]() {
#if HAVE_TDM
        if (wave == 0) __builtin_amdgcn_s_wait_tensorcnt(0);
#endif
    };

    // Q fragments for this wave's 16-row tile stay in registers (16 x 128)
    v16bf qf[4];
#pragma unroll
    for (int kk = 0; kk < 4; ++kk) qf[kk] = load_a_frag(Qp, Dc, q0, kk * 32);

    v8f o[8];
#pragma unroll
    for (int t = 0; t < 8; ++t) o[t] = {};
    float m_i[8], l_i[8];
#pragma unroll
    for (int r = 0; r < 8; ++r) { m_i[r] = -__builtin_inff(); l_i[r] = 0.f; }

    const float scale = 0.08838834764831845f;   // 1/sqrt(128)
    const int qmax = q0 + 15;                   // this wave's causal bound

    const int nblk = (qblk + 64 + 31) / 32;     // 32-key blocks for the block
    issue_tiles(0, 0);
    wait_tiles();
    __syncthreads();

    for (int ib = 0; ib < nblk; ++ib) {
        const int j0  = ib * 32;
        const int cur = ib & 1;
        if (ib + 1 < nblk) issue_tiles((ib + 1) * 32, cur ^ 1);

        if (j0 <= qmax) {   // wave-uniform: skip fully-masked key blocks
            // S = Q * K^T (two 16x16 score frags, K-dim 128 = 4 chained WMMAs)
            v8f sc[2];
#pragma unroll
            for (int hf = 0; hf < 2; ++hf) {
                v8f c = {};
#pragma unroll
                for (int kk = 0; kk < 4; ++kk) {
                    v16bf bk = load_b_frag(&Klds[cur][0], KROW, kk * 32, hf * 16);
                    c = wmma_bf16(qf[kk], bk, c);
                }
                sc[hf] = c;
            }

            // causal mask + online softmax (rows across 16 lanes of a half)
            float p0v[8], p1v[8], corr[8];
#pragma unroll
            for (int r = 0; r < 8; ++r) {
                const int qrow = q0 + r + 8 * halfl;
                float s0 = sc[0][r] * scale;
                float s1 = sc[1][r] * scale;
                if (j0 + nl      > qrow) s0 = -__builtin_inff();
                if (j0 + 16 + nl > qrow) s1 = -__builtin_inff();
                float mx = fmaxf(s0, s1);
#pragma unroll
                for (int off = 1; off < 16; off <<= 1)
                    mx = fmaxf(mx, __shfl_xor(mx, off, 32));
                const float mn = fmaxf(m_i[r], mx);
                const float cr = __expf(m_i[r] - mn);
                m_i[r] = mn;
                const float p0 = __expf(s0 - mn);
                const float p1 = __expf(s1 - mn);
                float rs = p0 + p1;
#pragma unroll
                for (int off = 1; off < 16; off <<= 1)
                    rs += __shfl_xor(rs, off, 32);
                l_i[r]  = l_i[r] * cr + rs;
                corr[r] = cr;
                p0v[r] = p0; p1v[r] = p1;
            }
#pragma unroll
            for (int t = 0; t < 8; ++t)
#pragma unroll
                for (int r = 0; r < 8; ++r) o[t][r] *= corr[r];

            // Re-shape P: C-fragment layout -> A-fragment layout via this
            // wave's private LDS slice (intra-wave ordering: s_wait_dscnt)
#pragma unroll
            for (int r = 0; r < 8; ++r) {
                Plds[wave][(r + 8 * halfl) * 32 + nl]      = f2bf(p0v[r]);
                Plds[wave][(r + 8 * halfl) * 32 + 16 + nl] = f2bf(p1v[r]);
            }
            asm volatile("s_wait_dscnt 0" ::: "memory");
            v16bf pf = load_a_frag((const unsigned short*)&Plds[wave][0], 32, 0, 0);

            // O += P * V   (16x32 times 32x128 -> 8 WMMAs)
#pragma unroll
            for (int t = 0; t < 8; ++t) {
                v16bf vf = load_b_frag(&Vlds[cur][0], VROW, 0, t * 16);
                o[t] = wmma_bf16(pf, vf, o[t]);
            }
        }

        if (ib + 1 < nblk) wait_tiles();
        __syncthreads();
    }

    // epilogue: normalize and store bf16
#pragma unroll
    for (int t = 0; t < 8; ++t) {
#pragma unroll
        for (int r = 0; r < 8; ++r) {
            const int qrow = q0 + r + 8 * halfl;
            const float v = o[t][r] / l_i[r];
            Ob[(size_t)(b * Sc + qrow) * Dc + h * HDc + t * 16 + nl] = f2bf(v);
        }
    }
}

// ---------------------------------------------------------------------------
// Launch: cast -> transpose weights -> QKV GEMMs -> flash attn -> out proj
// ---------------------------------------------------------------------------
extern "C" void kernel_launch(void* const* d_in, const int* in_sizes, int n_in,
                              void* d_out, int out_size, void* d_ws, size_t ws_size,
                              hipStream_t stream) {
    const float* x  = (const float*)d_in[0];
    const float* Wq = (const float*)d_in[1];
    const float* Wk = (const float*)d_in[2];
    const float* Wv = (const float*)d_in[3];
    const float* Wp = (const float*)d_in[4];
    float* out = (float*)d_out;

    const size_t MD = (size_t)Mc * Dc;   // 8388608 elems
    const size_t DD = (size_t)Dc * Dc;   // 4194304 elems

    unsigned short* p   = (unsigned short*)d_ws;
    unsigned short* xb  = p;  p += MD;
    unsigned short* WqT = p;  p += DD;
    unsigned short* WkT = p;  p += DD;
    unsigned short* WvT = p;  p += DD;
    unsigned short* WpT = p;  p += DD;
    unsigned short* Qb  = p;  p += MD;
    unsigned short* Ktb = p;  p += MD;
    unsigned short* Vbb = p;  p += MD;
    unsigned short* Ob  = p;  p += MD;   // total ~117 MB of d_ws

    // 1) cast activations
    cast_f32_to_bf16<<<(int)(MD / 4 / 256), 256, 0, stream>>>(x, xb, (int)MD);

    // 2) cast + transpose weights (W.T layout so GEMM B rows are contiguous)
    dim3 tb(32, 8), tg(Dc / 32, Dc / 32);
    transpose_cast<<<tg, tb, 0, stream>>>(Wq, WqT);
    transpose_cast<<<tg, tb, 0, stream>>>(Wk, WkT);
    transpose_cast<<<tg, tb, 0, stream>>>(Wv, WvT);
    transpose_cast<<<tg, tb, 0, stream>>>(Wp, WpT);

    // 3) QKV projections (K stored transposed per head for attention)
    dim3 gg(Dc / 64, Mc / 64);           // 64x64 tile per 4-wave block
    gemm_bf16<<<gg, 128, 0, stream>>>(xb, WqT, Qb,  0);
    gemm_bf16<<<gg, 128, 0, stream>>>(xb, WkT, Ktb, 1);
    gemm_bf16<<<gg, 128, 0, stream>>>(xb, WvT, Vbb, 0);

    // 4) causal flash attention (shared TDM double-buffered K^T / V tiles)
    flash_attn<<<dim3(Sc / 64, Bc * Hc), 128, 0, stream>>>(Qb, Ktb, Vbb, Ob);

    // 5) output projection, fp32 store
    gemm_bf16<<<gg, 128, 0, stream>>>(Ob, WpT, out, 2);
}
